// EagerRNN_12429635355005
// MI455X (gfx1250) — matrix operand order repaired and checked by reading, not compiled, not run on hardware
//
#include <hip/hip_runtime.h>
#include <math.h>

// ---------------------------------------------------------------------------
// EagerRNN on gfx1250: fused persistent RNN with bf16 WMMA (f32 accumulate).
//   state_{t+1} = tanh(b + x_t @ Wx + state_t @ Wh)
// Combined K over [x_t (F=512) ; state (H=1024)] against W^T rows.
// W^T slice staged in LDS once (time-invariant); B-frags via ds_load_b128.
// Hardware v_tanh_f32 epilogue; grid-wide atomic barrier per timestep.
// ---------------------------------------------------------------------------

typedef __bf16 bf16_t;
typedef __attribute__((ext_vector_type(16))) __bf16 v16bf;
typedef __attribute__((ext_vector_type(8)))  __bf16 v8bf;
typedef __attribute__((ext_vector_type(8)))  float  v8f;

#define RNN_B 64
#define RNN_T 512
#define RNN_F 512
#define RNN_H 1024
#define RNN_K (RNN_F + RNN_H)   // 1536

#define GRID_BLOCKS 32           // each block owns 32 columns of H
#define BLOCK_THREADS 128        // 4 wave32s -> rows 0..63 in 16-row strips
#define LDS_WT_ELEMS (32 * RNN_K)            // 49152 bf16 = 96 KiB
#define LDS_WT_BYTES (LDS_WT_ELEMS * 2)

// ---- workspace layout (bytes) ---------------------------------------------
#define WS_OFF_XBF   ((size_t)0)                         // 64*512*512*2 = 32 MiB
#define WS_OFF_WT    ((size_t)33554432)                  // 1024*1536*2 = 3 MiB
#define WS_OFF_S0    ((size_t)36700160)                  // 64*1024*2
#define WS_OFF_S1    ((size_t)36831232)                  // 64*1024*2
#define WS_OFF_CNT   ((size_t)36962304)                  // 4 B barrier counter

// ---------------------------------------------------------------------------
__global__ void rnn_cvt_x_kernel(const float* __restrict__ x,
                                 bf16_t* __restrict__ xbf, int n) {
  int i = blockIdx.x * blockDim.x + threadIdx.x;
  if (i < n) xbf[i] = (bf16_t)x[i];
}

// WT[n][j] = W[j][n], f32 -> bf16.  (reads coalesced over n)
__global__ void rnn_build_wt_kernel(const float* __restrict__ W,
                                    bf16_t* __restrict__ WT) {
  int i = blockIdx.x * blockDim.x + threadIdx.x;   // over RNN_H * RNN_K
  int n = i & (RNN_H - 1);
  int j = i >> 10;
  if (j < RNN_K) WT[(size_t)n * RNN_K + j] = (bf16_t)W[(size_t)j * RNN_H + n];
}

__global__ void rnn_zero_state_kernel(bf16_t* s0, bf16_t* s1, unsigned* cnt) {
  int i = blockIdx.x * blockDim.x + threadIdx.x;
  if (i < RNN_B * RNN_H) { s0[i] = (bf16_t)0.0f; s1[i] = (bf16_t)0.0f; }
  if (i == 0) *cnt = 0u;
}

// ---------------------------------------------------------------------------
__device__ __forceinline__ void rnn_grid_sync(unsigned* cnt, unsigned target) {
  __builtin_amdgcn_fence(__ATOMIC_RELEASE, "agent");
  __syncthreads();
  if (threadIdx.x == 0) {
    __hip_atomic_fetch_add(cnt, 1u, __ATOMIC_RELEASE, __HIP_MEMORY_SCOPE_AGENT);
    unsigned v;
    do {
      __builtin_amdgcn_s_sleep(8);
      v = __hip_atomic_load(cnt, __ATOMIC_RELAXED, __HIP_MEMORY_SCOPE_AGENT);
    } while (v < target);
  }
  __syncthreads();
  __builtin_amdgcn_fence(__ATOMIC_ACQUIRE, "agent");
}

__device__ __forceinline__ v16bf rnn_cat8(v8bf lo, v8bf hi) {
  return __builtin_shufflevector(lo, hi, 0,1,2,3,4,5,6,7,8,9,10,11,12,13,14,15);
}

// Branchless tanh: prefer the CDNA5 hardware transcendental (v_tanh_f32).
__device__ __forceinline__ float rnn_tanh(float x) {
#if __has_builtin(__builtin_amdgcn_tanhf)
  return __builtin_amdgcn_tanhf(x);
#else
  // tanh(x) = sign(x) * (1 - e) / (1 + e),  e = exp(-2|x|) = exp2(-2*log2e*|x|)
  float ax = __builtin_fabsf(x);
  float e  = __builtin_amdgcn_exp2f(ax * -2.8853900817779268f);
  float y  = (1.0f - e) * __builtin_amdgcn_rcpf(1.0f + e);
  return __builtin_copysignf(y, x);
#endif
}

// ---------------------------------------------------------------------------
__global__ __launch_bounds__(BLOCK_THREADS, 1)
void rnn_persistent_kernel(const bf16_t* __restrict__ xbf,
                           const bf16_t* __restrict__ WT,
                           const float*  __restrict__ bias,
                           bf16_t* s0, bf16_t* s1,
                           float* __restrict__ out,
                           unsigned* cnt) {
  extern __shared__ char smem_raw[];
  bf16_t* wt_lds = (bf16_t*)smem_raw;    // [32 local cols][RNN_K], 96 KiB

  const int lane = threadIdx.x & 31;
  const int wave = threadIdx.x >> 5;
  const int m0 = wave * 16;              // 16-row strip per wave (M = batch)
  const int n0 = blockIdx.x * 32;        // 32-column slice per block
  const int n1 = n0 + 16;

  const int nlo   = lane & 15;
  const int akoff = (lane >> 4) << 4;    // A-frag K sub-offset: 0 / 16
  const int bkoff = (lane >> 4) << 4;    // B-frag K sub-offset: 0 / 16
  const int mrow  = m0 + nlo;            // A row held by this lane

  // ---- stage this block's W^T slice (time-invariant) into LDS -------------
  {
    const bf16_t* src = WT + (size_t)n0 * RNN_K;   // 32 consecutive rows
    for (int off = threadIdx.x * 8; off < LDS_WT_ELEMS; off += BLOCK_THREADS * 8) {
      *(v8bf*)(wt_lds + off) = *(const v8bf*)(src + off);
    }
  }
  __syncthreads();

  const bf16_t* xrow_base = xbf + (size_t)mrow * (RNN_T * RNN_F);
  const bf16_t* bl0 = wt_lds + (size_t)nlo        * RNN_K + bkoff;  // tile n0
  const bf16_t* bl1 = wt_lds + (size_t)(16 + nlo) * RNN_K + bkoff;  // tile n1

  const float bias0 = bias[n0 + nlo];
  const float bias1 = bias[n1 + nlo];
  const int   mout  = m0 + ((lane >> 4) << 3);   // C/D row base for this lane

  for (int t = 0; t < RNN_T; ++t) {
    const bf16_t* ssrc = (t & 1) ? s1 : s0;
    bf16_t*       sdst = (t & 1) ? s0 : s1;

    const bf16_t* xrow = xrow_base + t * RNN_F + akoff;
    const bf16_t* srow = ssrc + (size_t)mrow * RNN_H + akoff;

    // Prefetch next timestep's x row while we compute this one.
    if (t + 1 < RNN_T)
      __builtin_prefetch(xrow_base + (size_t)(t + 1) * RNN_F + akoff, 0, 0);

    v8f acc0 = {0.f, 0.f, 0.f, 0.f, 0.f, 0.f, 0.f, 0.f};
    v8f acc1 = {0.f, 0.f, 0.f, 0.f, 0.f, 0.f, 0.f, 0.f};

    // ---- phase 1: K = 0..511 from x_t ------------------------------------
    #pragma unroll 4
    for (int k = 0; k < RNN_F; k += 32) {
      v16bf afrag = rnn_cat8(*(const v8bf*)(xrow + k),
                             *(const v8bf*)(xrow + k + 8));
      v16bf bf0 = rnn_cat8(*(const v8bf*)(bl0 + k), *(const v8bf*)(bl0 + k + 8));
      v16bf bf1 = rnn_cat8(*(const v8bf*)(bl1 + k), *(const v8bf*)(bl1 + k + 8));
      acc0 = __builtin_amdgcn_wmma_f32_16x16x32_bf16(
          false, afrag, false, bf0, (short)0, acc0, false, false);
      acc1 = __builtin_amdgcn_wmma_f32_16x16x32_bf16(
          false, afrag, false, bf1, (short)0, acc1, false, false);
    }

    // ---- phase 2: K = 512..1535 from recurrent state ---------------------
    #pragma unroll 4
    for (int k = 0; k < RNN_H; k += 32) {
      const int j = RNN_F + k;
      v16bf afrag = rnn_cat8(*(const v8bf*)(srow + k),
                             *(const v8bf*)(srow + k + 8));
      v16bf bf0 = rnn_cat8(*(const v8bf*)(bl0 + j), *(const v8bf*)(bl0 + j + 8));
      v16bf bf1 = rnn_cat8(*(const v8bf*)(bl1 + j), *(const v8bf*)(bl1 + j + 8));
      acc0 = __builtin_amdgcn_wmma_f32_16x16x32_bf16(
          false, afrag, false, bf0, (short)0, acc0, false, false);
      acc1 = __builtin_amdgcn_wmma_f32_16x16x32_bf16(
          false, afrag, false, bf1, (short)0, acc1, false, false);
    }

    // ---- epilogue: bias + hardware tanh, write next state (bf16) ---------
    float v0s[8], v1s[8];
    #pragma unroll
    for (int r = 0; r < 8; ++r) {
      v0s[r] = rnn_tanh(acc0[r] + bias0);
      v1s[r] = rnn_tanh(acc1[r] + bias1);
      const int m = mout + r;
      sdst[(size_t)m * RNN_H + n0 + nlo] = (bf16_t)v0s[r];
      sdst[(size_t)m * RNN_H + n1 + nlo] = (bf16_t)v1s[r];
    }
    if (t == RNN_T - 1) {
      #pragma unroll
      for (int r = 0; r < 8; ++r) {
        const int m = mout + r;
        out[(size_t)m * RNN_H + n0 + nlo] = v0s[r];
        out[(size_t)m * RNN_H + n1 + nlo] = v1s[r];
      }
    }

    rnn_grid_sync(cnt, (unsigned)gridDim.x * (unsigned)(t + 1));
  }
}

// ---------------------------------------------------------------------------
extern "C" void kernel_launch(void* const* d_in, const int* in_sizes, int n_in,
                              void* d_out, int out_size, void* d_ws, size_t ws_size,
                              hipStream_t stream) {
  const float* x = (const float*)d_in[0];   // (64, 512, 512) f32
  const float* W = (const float*)d_in[1];   // (1536, 1024) f32
  const float* b = (const float*)d_in[2];   // (1024,) f32
  float* out = (float*)d_out;               // (64, 1024) f32

  char* ws = (char*)d_ws;
  bf16_t*   xbf = (bf16_t*)(ws + WS_OFF_XBF);
  bf16_t*   WT  = (bf16_t*)(ws + WS_OFF_WT);
  bf16_t*   s0  = (bf16_t*)(ws + WS_OFF_S0);
  bf16_t*   s1  = (bf16_t*)(ws + WS_OFF_S1);
  unsigned* cnt = (unsigned*)(ws + WS_OFF_CNT);

  const int nx = RNN_B * RNN_T * RNN_F;
  rnn_cvt_x_kernel<<<(nx + 255) / 256, 256, 0, stream>>>(x, xbf, nx);

  const int nwt = RNN_H * RNN_K;
  rnn_build_wt_kernel<<<(nwt + 255) / 256, 256, 0, stream>>>(W, WT);

  rnn_zero_state_kernel<<<(RNN_B * RNN_H + 255) / 256, 256, 0, stream>>>(s0, s1, cnt);

  rnn_persistent_kernel<<<GRID_BLOCKS, BLOCK_THREADS, LDS_WT_BYTES, stream>>>(
      xbf, WT, b, s0, s1, out, cnt);
}
